// img_attention_59691455479959
// MI455X (gfx1250) — matrix operand order repaired and checked by reading, not compiled, not run on hardware
//
#include <hip/hip_runtime.h>

// ---------------------------------------------------------------------------
// img_attention on MI455X (gfx1250, wave32, WMMA)
// B=128, N=49, G=4096, H=1024, A=1024, M = B*N = 6272
// GEMMs run through v_wmma_f32_16x16x32_bf16 (f32 accumulate); f32 operands
// are converted to bf16 with native v_cvt during global->LDS staging, packed
// two-per-dword so LDS stores are b32 and fragment reads are ds_load_b128.
// ---------------------------------------------------------------------------

typedef __bf16 bf16_t;
typedef __attribute__((ext_vector_type(16))) __bf16 v16bf;
typedef __attribute__((ext_vector_type(8)))  float  v8f;

__device__ __forceinline__ unsigned int pack2bf(float a, float b) {
  unsigned short x = __builtin_bit_cast(unsigned short, (bf16_t)a);
  unsigned short y = __builtin_bit_cast(unsigned short, (bf16_t)b);
  return (unsigned int)x | ((unsigned int)y << 16);
}

// Tiled WMMA GEMM: C[M x Nfull] = A[M x K] * Bw[K x Nfull] (+ epilogue)
//   block tile 64(M) x 128(N), BK = 32, 256 threads = 8 waves (2 x 4),
//   each wave owns a 32x32 tile = 2x2 v_wmma_f32_16x16x32_bf16.
// EPI = 0 : Cf[m,n] = acc (+ bias[n])            (f32 store)
// EPI = 1 : Cbf[m,n] = bf16(acc + bias[n])       (bf16 store)
// EPI = 2 : attention epilogue ->
//           t = tanh(acc + sproj[m/49, n] + cov[m]*Wc[n]) * vvec[n]
//           part[m*gridDim.x + blockIdx.x] = sum_n t   (deterministic partials)
template <typename AT, int EPI>
__global__ __launch_bounds__(256)
void gemm_wmma_kernel(const AT* __restrict__ A, const float* __restrict__ Bw,
                      float* __restrict__ Cf, bf16_t* __restrict__ Cbf,
                      const float* __restrict__ bias,
                      const float* __restrict__ sproj,
                      const float* __restrict__ cov,
                      const float* __restrict__ Wc,
                      const float* __restrict__ vvec,
                      float* __restrict__ part,
                      int Nfull, int K) {
  __shared__ __align__(16) bf16_t As[64 * 32];   // [m][k]
  __shared__ __align__(16) bf16_t Bs[128 * 32];  // [n][k] (K contiguous)
  __shared__ float sPart[4][64];                 // EPI==2 cross-wave reduction

  unsigned int* As32 = (unsigned int*)As;
  unsigned int* Bs32 = (unsigned int*)Bs;

  const int tid   = threadIdx.x;
  const int lane  = tid & 31;
  const int wave  = tid >> 5;
  const int waveM = wave & 1;        // 2 waves along M
  const int waveN = wave >> 1;       // 4 waves along N
  const int bM    = blockIdx.y * 64;
  const int bN    = blockIdx.x * 128;

  v8f acc[2][2];
  {
    v8f z = {};
    acc[0][0] = z; acc[0][1] = z; acc[1][0] = z; acc[1][1] = z;
  }

  // staging assignments
  const int ar   = tid >> 2;           // 0..63  A row
  const int ac   = (tid & 3) << 3;     // 0,8,16,24 (8 consecutive K)
  const int krow = (tid >> 4) << 1;    // 0,2,..,30  B K-row pair
  const int nc0  = (tid & 15) << 3;    // 0..120 (8 consecutive N)

  const int mloc = lane & 15;
  const int kA   = (lane < 16) ? 0 : 8;    // A-frag K base (ISA 16-bit A layout)
  const int kB   = (lane < 16) ? 0 : 16;   // unused; B-frag reads use kA

  for (int kk = 0; kk < K; kk += 32) {
    __syncthreads();
    // ---- stage A tile ----
    const AT* ap = A + (size_t)(bM + ar) * K + kk + ac;
    if constexpr (sizeof(AT) == 4) {     // f32 source: convert + pack
#pragma unroll
      for (int j = 0; j < 4; ++j)
        As32[(ar * 32 + ac) / 2 + j] = pack2bf((float)ap[2 * j], (float)ap[2 * j + 1]);
    } else {                             // bf16 source: straight dword copy
      const unsigned int* ap32 = (const unsigned int*)ap;
#pragma unroll
      for (int j = 0; j < 4; ++j)
        As32[(ar * 32 + ac) / 2 + j] = ap32[j];
    }
    // ---- stage B tile transposed, two K rows per thread packed per dword ----
    const float* bp0 = Bw + (size_t)(kk + krow) * Nfull + bN + nc0;
    const float* bp1 = bp0 + Nfull;
#pragma unroll
    for (int i = 0; i < 8; ++i)
      Bs32[((nc0 + i) * 32 + krow) / 2] = pack2bf(bp0[i], bp1[i]);
    if (kk + 32 < K) {                   // global_prefetch_b8 of next K tile
      __builtin_prefetch(ap + 32, 0, 1);
      __builtin_prefetch(bp0 + (size_t)32 * Nfull, 0, 1);
    }
    __syncthreads();

    // ---- compute 2x2 WMMA per wave ----
#pragma unroll
    for (int mi = 0; mi < 2; ++mi) {
      v16bf af;
      const int mrow = waveM * 32 + mi * 16 + mloc;
#pragma unroll
      for (int p = 0; p < 8; ++p) {
        const int kp = (p < 4) ? (kA + 2 * p) : (kA + 8 + 2 * p);
        af[2 * p]     = As[mrow * 32 + kp];
        af[2 * p + 1] = As[mrow * 32 + kp + 1];
      }
#pragma unroll
      for (int ni = 0; ni < 2; ++ni) {
        v16bf bfv;
        const int ncol = waveN * 32 + ni * 16 + mloc;
#pragma unroll
        for (int i = 0; i < 16; ++i) bfv[i] = Bs[ncol * 32 + kA + i + ((i & 8) ? 8 : 0)];
        acc[mi][ni] = __builtin_amdgcn_wmma_f32_16x16x32_bf16(
            false, af, false, bfv, (short)0, acc[mi][ni], false, false);
      }
    }
  }

  // ---- epilogue (C/D layout: VGPR v -> M = v + 8*(lane>=16), N = lane&15) ----
  const int colL    = lane & 15;
  const int rowHalf = (lane >> 4) << 3;

  if constexpr (EPI == 2) {
    float rowsum[2][8];
#pragma unroll
    for (int mi = 0; mi < 2; ++mi)
#pragma unroll
      for (int v8 = 0; v8 < 8; ++v8) rowsum[mi][v8] = 0.0f;

#pragma unroll
    for (int mi = 0; mi < 2; ++mi)
#pragma unroll
      for (int ni = 0; ni < 2; ++ni)
#pragma unroll
        for (int v8 = 0; v8 < 8; ++v8) {
          const int m = bM + waveM * 32 + mi * 16 + rowHalf + v8;
          const int n = bN + waveN * 32 + ni * 16 + colL;
          const int b = m / 49;
          float x = acc[mi][ni][v8] + sproj[(size_t)b * Nfull + n] + cov[m] * Wc[n];
          rowsum[mi][v8] += tanhf(x) * vvec[n];
        }
    // reduce over the 16 columns held by this half-wave
#pragma unroll
    for (int mi = 0; mi < 2; ++mi)
#pragma unroll
      for (int v8 = 0; v8 < 8; ++v8) {
        float s = rowsum[mi][v8];
        s += __shfl_xor(s, 1, 32);
        s += __shfl_xor(s, 2, 32);
        s += __shfl_xor(s, 4, 32);
        s += __shfl_xor(s, 8, 32);
        rowsum[mi][v8] = s;
      }
    if (colL == 0) {
#pragma unroll
      for (int mi = 0; mi < 2; ++mi)
#pragma unroll
        for (int v8 = 0; v8 < 8; ++v8)
          sPart[waveN][waveM * 32 + mi * 16 + rowHalf + v8] = rowsum[mi][v8];
    }
    __syncthreads();
    if (tid < 64) {
      const float s = sPart[0][tid] + sPart[1][tid] + sPart[2][tid] + sPart[3][tid];
      part[(size_t)(bM + tid) * gridDim.x + blockIdx.x] = s;
    }
  } else {
#pragma unroll
    for (int mi = 0; mi < 2; ++mi)
#pragma unroll
      for (int ni = 0; ni < 2; ++ni)
#pragma unroll
        for (int v8 = 0; v8 < 8; ++v8) {
          const int m = bM + waveM * 32 + mi * 16 + rowHalf + v8;
          const int n = bN + waveN * 32 + ni * 16 + colL;
          float x = acc[mi][ni][v8];
          if (bias) x += bias[n];
          if constexpr (EPI == 1) Cbf[(size_t)m * Nfull + n] = (bf16_t)x;
          else                    Cf[(size_t)m * Nfull + n]  = x;
        }
  }
}

// softmax over N=49 regions; scores come as 8 column-block partials per row
__global__ void softmax49_kernel(const float* __restrict__ part,
                                 float* __restrict__ alpha) {
  const int b = threadIdx.x;  // 128 batches, one block of 128 threads
  if (b >= 128) return;
  float sc[49];
  float mx = -1e30f;
#pragma unroll 1
  for (int n = 0; n < 49; ++n) {
    const float* p = part + (size_t)(b * 49 + n) * 8;
    float s = 0.0f;
#pragma unroll
    for (int j = 0; j < 8; ++j) s += p[j];
    sc[n] = s;
    mx = fmaxf(mx, s);
  }
  float den = 0.0f;
#pragma unroll 1
  for (int n = 0; n < 49; ++n) { sc[n] = expf(sc[n] - mx); den += sc[n]; }
  const float inv = 1.0f / den;
#pragma unroll 1
  for (int n = 0; n < 49; ++n) alpha[b * 49 + n] = sc[n] * inv;
}

// c_img[b,h] = sum_n alpha[b,n] * g_star[b,n,h]  (H=1024, 256 thr x 4 floats)
__global__ __launch_bounds__(256)
void context_kernel(const float* __restrict__ alpha,
                    const float* __restrict__ gstar,
                    float* __restrict__ out) {
  __shared__ float aS[49];
  const int b   = blockIdx.x;
  const int tid = threadIdx.x;
  if (tid < 49) aS[tid] = alpha[b * 49 + tid];
  __syncthreads();
  const int h = tid * 4;
  float a0 = 0.f, a1 = 0.f, a2 = 0.f, a3 = 0.f;
#pragma unroll 1
  for (int n = 0; n < 49; ++n) {
    const float w = aS[n];
    const float4 g = *(const float4*)(gstar + (size_t)(b * 49 + n) * 1024 + h);
    a0 += w * g.x; a1 += w * g.y; a2 += w * g.z; a3 += w * g.w;
  }
  float4 r; r.x = a0; r.y = a1; r.z = a2; r.w = a3;
  *(float4*)(out + (size_t)b * 1024 + h) = r;
}

extern "C" void kernel_launch(void* const* d_in, const int* in_sizes, int n_in,
                              void* d_out, int out_size, void* d_ws, size_t ws_size,
                              hipStream_t stream) {
  const float* gf   = (const float*)d_in[0];   // [128,49,4096]
  const float* s_t  = (const float*)d_in[1];   // [128,1024]
  const float* cov  = (const float*)d_in[2];   // [128,49,1]
  const float* W_g  = (const float*)d_in[3];   // [4096,4096]
  const float* b_g  = (const float*)d_in[4];   // [4096]
  const float* W_gs = (const float*)d_in[5];   // [4096,1024]
  const float* b_gs = (const float*)d_in[6];   // [1024]
  const float* W_h  = (const float*)d_in[7];   // [1024,1024]
  const float* W_s  = (const float*)d_in[8];   // [1024,1024]
  const float* W_c  = (const float*)d_in[9];   // [1,1024]
  const float* vv   = (const float*)d_in[10];  // [1024]
  float* out = (float*)d_out;

  const int Bb = 128, Gd = 4096, Hd = 1024, Ad = 1024;
  const int M = Bb * 49;  // 6272

  char* w = (char*)d_ws;
  auto take = [&](size_t bytes) -> void* {
    void* p = (void*)w;
    w += (bytes + 255) & ~(size_t)255;
    return p;
  };
  bf16_t* g1    = (bf16_t*)take((size_t)M * Gd * sizeof(bf16_t)); // 51.4 MB
  float*  gstar = (float*) take((size_t)M * Hd * sizeof(float));  // 25.7 MB
  float*  sproj = (float*) take((size_t)Bb * Ad * sizeof(float));
  float*  part  = (float*) take((size_t)M * 8 * sizeof(float));
  float*  alpha = (float*) take((size_t)M * sizeof(float));

  const dim3 blk(256);

  // s_proj = s_t @ W_s               [128,1024]
  gemm_wmma_kernel<float, 0><<<dim3(Ad / 128, Bb / 64), blk, 0, stream>>>(
      s_t, W_s, sproj, nullptr, nullptr,
      nullptr, nullptr, nullptr, nullptr, nullptr, Ad, Hd);

  // g1 = X @ W_g + b_g (bf16 store)  [6272,4096]
  gemm_wmma_kernel<float, 1><<<dim3(Gd / 128, M / 64), blk, 0, stream>>>(
      gf, W_g, nullptr, g1, b_g,
      nullptr, nullptr, nullptr, nullptr, nullptr, Gd, Gd);

  // g_star = g1 @ W_gs + b_gs        [6272,1024] f32
  gemm_wmma_kernel<bf16_t, 0><<<dim3(Hd / 128, M / 64), blk, 0, stream>>>(
      g1, W_gs, gstar, nullptr, b_gs,
      nullptr, nullptr, nullptr, nullptr, nullptr, Hd, Gd);

  // score partials: tanh(g_star@W_h + sproj + cov*W_c) . v  -> part[m, 8]
  gemm_wmma_kernel<float, 2><<<dim3(Ad / 128, M / 64), blk, 0, stream>>>(
      gstar, W_h, nullptr, nullptr, nullptr,
      sproj, cov, W_c, vv, part, Ad, Hd);

  softmax49_kernel<<<1, 128, 0, stream>>>(part, alpha);
  context_kernel<<<Bb, 256, 0, stream>>>(alpha, gstar, out);
}
